// SAE_20598663152228
// MI455X (gfx1250) — matrix operand order repaired and checked by reading, not compiled, not run on hardware
//
#include <hip/hip_runtime.h>

#define D_IN   768
#define D_SAE  24576
#define BATCH  2048
#define KSEL   64

// Encoder GEMM tiling
#define MB        64     // rows (M) per block
#define NTILE     128    // cols (N) per inner tile
#define NBLKCOLS  1536   // cols swept per block (12 tiles)
#define KT        32     // K per WMMA step (bf16 16x16x32)
#define KITERS    (D_IN / KT)            // 24
#define NTILES    (NBLKCOLS / NTILE)     // 12
#define NSTEPS    (NTILES * KITERS)      // 288
#define LDA       776    // padded LDS stride for A (halfwords), 16B-aligned rows
#define LDB       40     // padded LDS stride for B (halfwords), 16B-aligned rows

#define GLOBAL_AS __attribute__((address_space(1)))
#define LDS_AS    __attribute__((address_space(3)))

typedef __attribute__((ext_vector_type(16))) __bf16          v16bf;
typedef __attribute__((ext_vector_type(8)))  float           v8f;
typedef __attribute__((ext_vector_type(8)))  unsigned short  ush8;
typedef __attribute__((ext_vector_type(4)))  int             v4i;

union Frag {
  v16bf v;
  ush8  h[2];
};

__device__ inline unsigned short f2b(float f) {
  unsigned u = __builtin_bit_cast(unsigned, f);
  unsigned r = u + 0x7FFFu + ((u >> 16) & 1u);
  return (unsigned short)(r >> 16);
}

// Pack two f32 -> {lo, hi} bf16 in one u32 (v_cvt_pk_bf16_f32 when available).
__device__ inline unsigned pack_bf16(float lo, float hi) {
#if __has_builtin(__builtin_amdgcn_cvt_pk_bf16_f32)
  auto p = __builtin_amdgcn_cvt_pk_bf16_f32(lo, hi);
  return __builtin_bit_cast(unsigned, p);
#else
  return (unsigned)f2b(lo) | ((unsigned)f2b(hi) << 16);
#endif
}

// Stage one B tile: W_enc[kt*32 .. +32)[nbase .. +128) transposed to buf[n][k],
// two K-rows packed per u32 store. 512 items (4 u32 writes each) / 256 threads = 2 fixed iters.
__device__ inline void stage_B(const float* __restrict__ W_enc, int nc0, int nt, int kt,
                               unsigned short* __restrict__ buf, int tid)
{
  const int nbase = nc0 + nt * NTILE;
#pragma unroll
  for (int i = 0; i < ((KT / 2) * (NTILE / 4)) / 256; ++i) {  // 2 iterations
    int id  = tid + i * 256;
    int kk2 = id / (NTILE / 4);  // 0..15, pair of K rows (2*kk2, 2*kk2+1)
    int nq  = id % (NTILE / 4);  // 0..31, group of 4 N columns
    const float* s0 = W_enc + (size_t)(kt * KT + 2 * kk2) * D_SAE + nbase + nq * 4;
    float4 w0 = *(const float4*)s0;
    float4 w1 = *(const float4*)(s0 + D_SAE);
    unsigned short* base = buf + (size_t)(nq * 4) * LDB + 2 * kk2;
    *(unsigned*)(base + 0 * LDB) = pack_bf16(w0.x, w1.x);
    *(unsigned*)(base + 1 * LDB) = pack_bf16(w0.y, w1.y);
    *(unsigned*)(base + 2 * LDB) = pack_bf16(w0.z, w1.z);
    *(unsigned*)(base + 3 * LDB) = pack_bf16(w0.w, w1.w);
    if (kt + 1 < KITERS)
      __builtin_prefetch(s0 + (size_t)KT * D_SAE, 0, 3);  // near-cache prefetch of next K-slab
  }
}

// -------------------------------------------------------------------------
// Kernel 1: pre = (x - b_dec) @ W_enc + b_enc via bf16 WMMA, double-buffered B.
// -------------------------------------------------------------------------
__global__ __launch_bounds__(256) void sae_encoder_wmma(
    const float* __restrict__ x, const float* __restrict__ W_enc,
    const float* __restrict__ b_enc, const float* __restrict__ b_dec,
    float* __restrict__ pre)
{
  extern __shared__ unsigned short smem[];
  unsigned short* lA  = smem;                 // MB x LDA
  unsigned short* lB0 = smem + MB * LDA;      // NTILE x LDB (double buffered)
  unsigned short* lB1 = lB0 + NTILE * LDB;

  const int tid = threadIdx.x;
  const int m0  = blockIdx.y * MB;
  const int nc0 = blockIdx.x * NBLKCOLS;

  // Stage A once: 64 rows x 768 cols of (x - b_dec), f32 -> bf16 packed stores.
#pragma unroll 4
  for (int i = 0; i < (MB * (D_IN / 4)) / 256; ++i) {  // 48 iterations
    int id = tid + i * 256;
    int m  = id / (D_IN / 4);
    int kq = id % (D_IN / 4);
    float4 xv = *(const float4*)(x + (size_t)(m0 + m) * D_IN + kq * 4);
    float4 bd = *(const float4*)(b_dec + kq * 4);
    uint2 p;
    p.x = pack_bf16(xv.x - bd.x, xv.y - bd.y);
    p.y = pack_bf16(xv.z - bd.z, xv.w - bd.w);
    *(uint2*)(lA + (size_t)m * LDA + kq * 4) = p;
  }

  const int wave = tid >> 5;
  const int lane = tid & 31;
  const int msub = wave & 3;   // 16-row M sub-tile
  const int nsub = wave >> 2;  // 64-col N group
  const int lgrp = lane >> 4;
  const int lmod = lane & 15;

  stage_B(W_enc, nc0, 0, 0, lB0, tid);  // prologue: stage step 0

  v8f acc[4] = {};
  for (int s = 0; s < NSTEPS; ++s) {
    const int kt = s % KITERS;
    const int nt = s / KITERS;
    __syncthreads();  // staged buffer for step s visible; step s-1 reads retired

    if (s + 1 < NSTEPS)
      stage_B(W_enc, nc0, (s + 1) / KITERS, (s + 1) % KITERS,
              ((s + 1) & 1) ? lB1 : lB0, tid);

    const unsigned short* curB = (s & 1) ? lB1 : lB0;

    // A fragment: lane row M = msub*16 + lmod; element e holds
    // K = kt*32 + e + (e>=8 ? 8 : 0) + 8*lgrp -> two 16B chunks.
    Frag fa;
    const unsigned short* arow = lA + (size_t)(msub * 16 + lmod) * LDA + kt * KT + 8 * lgrp;
    fa.h[0] = *(const ush8*)(arow);
    fa.h[1] = *(const ush8*)(arow + 16);

    // B fragments: lane col N = nsub*64 + a*16 + lmod; elements e hold
    // K = kt*32 + e + (e>=8 ? 8 : 0) + 8*lgrp -> two 16B chunks.
    Frag fb[4];
#pragma unroll
    for (int a = 0; a < 4; ++a) {
      const unsigned short* brow = curB + (size_t)(nsub * 64 + a * 16 + lmod) * LDB + 8 * lgrp;
      fb[a].h[0] = *(const ush8*)(brow);
      fb[a].h[1] = *(const ush8*)(brow + 16);
    }
#pragma unroll
    for (int a = 0; a < 4; ++a)
      acc[a] = __builtin_amdgcn_wmma_f32_16x16x32_bf16(
          false, fa.v, false, fb[a].v, (short)0, acc[a], false, false);

    if (kt == KITERS - 1) {
      // C layout: element r -> M = r + 8*lgrp, N = lmod.
      const int nbase = nc0 + nt * NTILE;
#pragma unroll
      for (int a = 0; a < 4; ++a) {
        int n = nbase + nsub * 64 + a * 16 + lmod;
        float be = b_enc[n];
#pragma unroll
        for (int r = 0; r < 8; ++r) {
          int m = m0 + msub * 16 + r + 8 * lgrp;
          pre[(size_t)m * D_SAE + n] = acc[a][r] + be;
        }
        acc[a] = v8f{};
      }
    }
  }
}

// -------------------------------------------------------------------------
// Kernel 2: exact per-row top-64 via 4-pass radix select on order-keys.
// Row staged into LDS via CDNA5 async global->LDS DMA when available.
// -------------------------------------------------------------------------
__device__ inline unsigned fkey(float f) {
  unsigned u = __builtin_bit_cast(unsigned, f);
  return (u & 0x80000000u) ? ~u : (u | 0x80000000u);
}

__global__ __launch_bounds__(256) void sae_topk(
    const float* __restrict__ pre, float* __restrict__ vals, int* __restrict__ idxs)
{
  extern __shared__ float rowv[];  // D_SAE floats (96 KB)
  __shared__ unsigned hist[256];
  __shared__ unsigned s_prefix, s_rem;
  __shared__ unsigned cnt_gt, cnt_eq;

  const int r   = blockIdx.x;
  const int tid = threadIdx.x;
  const float* src = pre + (size_t)r * D_SAE;

#if __has_builtin(__builtin_amdgcn_global_load_async_to_lds_b128) && \
    __has_builtin(__builtin_amdgcn_s_wait_asynccnt)
  // Async DMA: each lane copies 16B chunks directly into LDS (no VGPR round-trip).
#pragma unroll
  for (int i = 0; i < D_SAE / (256 * 4); ++i) {  // 24 chunks per thread
    int off = (tid + i * 256) * 4;               // float index, 16B aligned
    __builtin_amdgcn_global_load_async_to_lds_b128(
        (GLOBAL_AS v4i*)(src + off), (LDS_AS v4i*)(rowv + off), 0, 0);
  }
  __builtin_amdgcn_s_wait_asynccnt(0);
#else
  for (int i = tid; i < D_SAE; i += 256) rowv[i] = src[i];
#endif
  if (tid == 0) { cnt_gt = 0; cnt_eq = 0; }

  unsigned prefix = 0, rem = KSEL;
  for (int p = 0; p < 4; ++p) {
    const int shift = 24 - 8 * p;
    hist[tid] = 0;
    __syncthreads();
    for (int i = tid; i < D_SAE; i += 256) {
      unsigned k = fkey(rowv[i]);
      if (p == 0 || (k >> (shift + 8)) == prefix)
        atomicAdd(&hist[(k >> shift) & 255u], 1u);
    }
    __syncthreads();
    if (tid == 0) {
      unsigned c = 0; int b = 255;
      for (; b > 0; --b) { c += hist[b]; if (c >= rem) break; }
      if (c < rem) c += hist[0];
      s_rem    = rem - (c - hist[b]);
      s_prefix = (prefix << 8) | (unsigned)b;
    }
    __syncthreads();
    prefix = s_prefix; rem = s_rem;
    __syncthreads();
  }

  const unsigned T = prefix;      // exact key of the 64th largest
  const unsigned G = KSEL - rem;  // count strictly greater than T
  float* vo = vals + (size_t)r * KSEL;
  int*   io = idxs + (size_t)r * KSEL;
  for (int i = tid; i < D_SAE; i += 256) {
    float f = rowv[i];
    unsigned k = fkey(f);
    if (k > T) {
      unsigned s = atomicAdd(&cnt_gt, 1u);
      vo[s] = fmaxf(f, 0.0f); io[s] = i;
    } else if (k == T) {
      unsigned q = atomicAdd(&cnt_eq, 1u);
      if (q < rem) { vo[G + q] = fmaxf(f, 0.0f); io[G + q] = i; }
    }
  }
}

// -------------------------------------------------------------------------
// Kernel 3: sparse decode + per-row squared error.
// -------------------------------------------------------------------------
__global__ __launch_bounds__(256) void sae_decode(
    const float* __restrict__ x, const float* __restrict__ W_dec,
    const float* __restrict__ b_dec, const float* __restrict__ vals,
    const int* __restrict__ idxs, float* __restrict__ partial)
{
  __shared__ float sv[KSEL];
  __shared__ int   si[KSEL];
  __shared__ float red[256];
  const int r   = blockIdx.x;
  const int tid = threadIdx.x;
  if (tid < KSEL) {
    sv[tid] = vals[(size_t)r * KSEL + tid];
    si[tid] = idxs[(size_t)r * KSEL + tid];
  }
  __syncthreads();

  const int c0 = tid, c1 = tid + 256, c2 = tid + 512;  // 768 = 3*256
  const float* xr = x + (size_t)r * D_IN;
  float a0 = b_dec[c0] - xr[c0];
  float a1 = b_dec[c1] - xr[c1];
  float a2 = b_dec[c2] - xr[c2];
  for (int j = 0; j < KSEL; ++j) {
    float v = sv[j];  // uniform across block
    if (v != 0.0f) {
      const float* wr = W_dec + (size_t)si[j] * D_IN;
      a0 += v * wr[c0];
      a1 += v * wr[c1];
      a2 += v * wr[c2];
    }
  }
  red[tid] = a0 * a0 + a1 * a1 + a2 * a2;
  __syncthreads();
  for (int s = 128; s > 0; s >>= 1) {
    if (tid < s) red[tid] += red[tid + s];
    __syncthreads();
  }
  if (tid == 0) partial[r] = red[0];
}

// -------------------------------------------------------------------------
// Kernel 4: deterministic final reduction.
// -------------------------------------------------------------------------
__global__ __launch_bounds__(256) void sae_reduce(
    const float* __restrict__ partial, float* __restrict__ out)
{
  __shared__ float red[256];
  float s = 0.0f;
  for (int i = threadIdx.x; i < BATCH; i += 256) s += partial[i];
  red[threadIdx.x] = s;
  __syncthreads();
  for (int t = 128; t > 0; t >>= 1) {
    if (threadIdx.x < t) red[threadIdx.x] += red[threadIdx.x + t];
    __syncthreads();
  }
  if (threadIdx.x == 0) out[0] = red[0];
}

// -------------------------------------------------------------------------
extern "C" void kernel_launch(void* const* d_in, const int* in_sizes, int n_in,
                              void* d_out, int out_size, void* d_ws, size_t ws_size,
                              hipStream_t stream) {
  (void)in_sizes; (void)n_in; (void)out_size; (void)ws_size;
  const float* x     = (const float*)d_in[0];
  const float* W_enc = (const float*)d_in[1];
  const float* W_dec = (const float*)d_in[2];
  const float* b_enc = (const float*)d_in[3];
  const float* b_dec = (const float*)d_in[4];
  // d_in[5] is k (== 64), hardcoded as KSEL.
  float* out = (float*)d_out;

  // Workspace layout
  float* pre     = (float*)d_ws;                           // BATCH*D_SAE f32 (201.3 MB)
  float* vals    = pre + (size_t)BATCH * D_SAE;            // BATCH*64 f32
  int*   idxs    = (int*)(vals + (size_t)BATCH * KSEL);    // BATCH*64 i32
  float* partial = (float*)(idxs + (size_t)BATCH * KSEL);  // BATCH f32

  dim3 egrid(D_SAE / NBLKCOLS, BATCH / MB);  // (16, 32) = 512 blocks
  size_t enc_lds = (size_t)(MB * LDA + 2 * NTILE * LDB) * sizeof(unsigned short);  // ~117 KB
  sae_encoder_wmma<<<egrid, dim3(256), enc_lds, stream>>>(x, W_enc, b_enc, b_dec, pre);

  size_t tk_lds = (size_t)D_SAE * sizeof(float);  // 96 KB
  sae_topk<<<dim3(BATCH), dim3(256), tk_lds, stream>>>(pre, vals, idxs);

  sae_decode<<<dim3(BATCH), dim3(256), 0, stream>>>(x, W_dec, b_dec, vals, idxs, partial);

  sae_reduce<<<dim3(1), dim3(256), 0, stream>>>(partial, out);
}
